// RAFT_62912680951984
// MI455X (gfx1250) — matrix operand order repaired and checked by reading, not compiled, not run on hardware
//
#include <hip/hip_runtime.h>
#include <hip/hip_bf16.h>

typedef __attribute__((ext_vector_type(16))) _Float16 v16h;
typedef __attribute__((ext_vector_type(8)))  _Float16 v8h;
typedef __attribute__((ext_vector_type(8)))  float    v8f;

#define BN_SCALE_F 0.99999500003749969f /* 1/sqrt(1+1e-5) */

static __device__ __forceinline__ v8h zero_v8h() {
    v8h z;
#pragma unroll
    for (int e = 0; e < 8; ++e) z[e] = (_Float16)0.f;
    return z;
}

// ---------------------------------------------------------------------------
// Weight pack: f32 [M][K] -> f16 [Mp][Kp], zero padded
// ---------------------------------------------------------------------------
__global__ void k_pack_w(const float* __restrict__ w, _Float16* __restrict__ A,
                         int M, int K, int Mp, int Kp) {
    long idx = (long)blockIdx.x * blockDim.x + threadIdx.x;
    long total = (long)Mp * Kp;
    if (idx >= total) return;
    int k = (int)(idx % Kp);
    int m = (int)(idx / Kp);
    float v = (m < M && k < K) ? w[(long)m * K + k] : 0.f;
    A[idx] = (_Float16)v;
}

// ---------------------------------------------------------------------------
// Feature pack for correlation: f32 [D][S] -> f16 [S][D]
// ---------------------------------------------------------------------------
__global__ void k_pack_feat(const float* __restrict__ in, _Float16* __restrict__ out,
                            int S, int D) {
    long idx = (long)blockIdx.x * blockDim.x + threadIdx.x;
    long total = (long)S * D;
    if (idx >= total) return;
    int k = (int)(idx % D);
    long s = idx / D;
    out[idx] = (_Float16)in[(long)k * S + s];
}

// ---------------------------------------------------------------------------
// im2col (one image): NCHW f32 -> Bt f16 [Hout*Wout][Kp], K-padding zeroed
// ---------------------------------------------------------------------------
__global__ void k_im2col(const float* __restrict__ in, int Cin, int Hin, int Win,
                         int KH, int KW, int stride, int padH, int padW,
                         int Hout, int Wout, _Float16* __restrict__ bt, int Kp) {
    long idx = (long)blockIdx.x * blockDim.x + threadIdx.x;
    long total = (long)Hout * Wout * Kp;
    if (idx >= total) return;
    int k = (int)(idx % Kp);
    long row = idx / Kp;
    float v = 0.f;
    int K = Cin * KH * KW;
    if (k < K) {
        int ci = k / (KH * KW);
        int r  = k - ci * (KH * KW);
        int ky = r / KW, kx = r - ky * KW;
        int ow = (int)(row % Wout);
        int oh = (int)(row / Wout);
        int ih = oh * stride - padH + ky;
        int iw = ow * stride - padW + kx;
        if (ih >= 0 && ih < Hin && iw >= 0 && iw < Win)
            v = in[((long)ci * Hin + ih) * Win + iw];
    }
    bt[idx] = (_Float16)v;
}

// ---------------------------------------------------------------------------
// WMMA GEMM: C[M][N] = alpha * A[Mp][Kp](f16) x Bt[N][Kp](f16)^T + bias[m]
// Block: 256 threads = 8 waves (2 M x 4 N), block tile 64x128, wave tile 32x32.
// A and B fragments both use contiguous k = 16*(lane/16) + s (wave32).
// ---------------------------------------------------------------------------
__global__ void __launch_bounds__(256)
k_gemm(const _Float16* __restrict__ A, const _Float16* __restrict__ Bt,
       float* __restrict__ Cc, const float* __restrict__ bias,
       int Mp, int Kp, int Nsp, int Mtrue, float alpha, int relu) {
    __shared__ __align__(64) _Float16 sA[64 * 32];
    __shared__ __align__(64) _Float16 sB[128 * 32];
    const int tid  = threadIdx.x;
    const int lane = tid & 31;
    const int wave = tid >> 5;
    const int wm = wave >> 2;  // 0..1
    const int wn = wave & 3;   // 0..3
    const int m0 = blockIdx.y * 64;
    const int n0 = blockIdx.x * 128;

    v8f acc[2][2];
#pragma unroll
    for (int i = 0; i < 2; ++i)
#pragma unroll
        for (int j = 0; j < 2; ++j)
#pragma unroll
            for (int e = 0; e < 8; ++e) acc[i][j][e] = 0.f;

    const int arow = tid >> 2;        // 0..63
    const int aseg = (tid & 3) << 3;  // 0,8,16,24 halves
    const int brow = tid >> 1;        // 0..127
    const int bseg = (tid & 1) << 4;  // 0,16 halves

    for (int k0 = 0; k0 < Kp; k0 += 32) {
        v8h av = zero_v8h();
        {
            int gr = m0 + arow;
            if (gr < Mp) av = *(const v8h*)(A + (size_t)gr * Kp + k0 + aseg);
        }
        v8h bv0 = zero_v8h(), bv1 = zero_v8h();
        {
            int gr = n0 + brow;
            if (gr < Nsp) {
                const _Float16* p = Bt + (size_t)gr * Kp + k0 + bseg;
                bv0 = *(const v8h*)p;
                bv1 = *(const v8h*)(p + 8);
            }
        }
        *(v8h*)(sA + arow * 32 + aseg) = av;
        *(v8h*)(sB + brow * 32 + bseg) = bv0;
        *(v8h*)(sB + brow * 32 + bseg + 8) = bv1;
        __syncthreads();

        const int kbA = (lane >> 4) << 4;   // 0 or 16
        const int kbB = (lane >> 4) << 4;   // 0 or 16
        const int rA  = (wm << 5) + (lane & 15);
        const int cB  = (wn << 5) + (lane & 15);

        v16h aF[2], bF[2];
#pragma unroll
        for (int i = 0; i < 2; ++i) {
            const _Float16* pa = sA + (rA + i * 16) * 32;
            v8h lo = *(const v8h*)(pa + kbA);
            v8h hi = *(const v8h*)(pa + kbA + 8);
#pragma unroll
            for (int e = 0; e < 8; ++e) { aF[i][e] = lo[e]; aF[i][e + 8] = hi[e]; }
        }
#pragma unroll
        for (int j = 0; j < 2; ++j)
            bF[j] = *(const v16h*)(sB + (cB + j * 16) * 32 + kbB);

#pragma unroll
        for (int i = 0; i < 2; ++i)
#pragma unroll
            for (int j = 0; j < 2; ++j)
                acc[i][j] = __builtin_amdgcn_wmma_f32_16x16x32_f16(
                    false, aF[i], false, bF[j], (short)0, acc[i][j], false, false);
        __syncthreads();
    }

#pragma unroll
    for (int i = 0; i < 2; ++i)
#pragma unroll
        for (int j = 0; j < 2; ++j) {
            int n  = n0 + (wn << 5) + j * 16 + (lane & 15);
            int mb = m0 + (wm << 5) + i * 16 + ((lane >> 4) << 3);
            if (n < Nsp) {
#pragma unroll
                for (int r = 0; r < 8; ++r) {
                    int m = mb + r;
                    if (m < Mtrue) {
                        float v = acc[i][j][r] * alpha + (bias ? bias[m] : 0.f);
                        if (relu) v = fmaxf(v, 0.f);
                        Cc[(size_t)m * Nsp + n] = v;
                    }
                }
            }
        }
}

// ---------------------------------------------------------------------------
// Instance norm (+optional relu), in place. grid = (C, B)
// ---------------------------------------------------------------------------
__global__ void __launch_bounds__(256)
k_instnorm(float* x, long imgStride, int S, int relu) {
    float* p = x + (long)blockIdx.y * imgStride + (long)blockIdx.x * S;
    float s = 0.f, ss = 0.f;
    for (int i = threadIdx.x; i < S; i += 256) { float v = p[i]; s += v; ss += v * v; }
    __shared__ float shs[256], shq[256];
    shs[threadIdx.x] = s; shq[threadIdx.x] = ss;
    __syncthreads();
    for (int o = 128; o > 0; o >>= 1) {
        if (threadIdx.x < o) { shs[threadIdx.x] += shs[threadIdx.x + o]; shq[threadIdx.x] += shq[threadIdx.x + o]; }
        __syncthreads();
    }
    float mean = shs[0] / (float)S;
    float var  = shq[0] / (float)S - mean * mean;
    float inv  = rsqrtf(fmaxf(var, 0.f) + 1e-5f);
    for (int i = threadIdx.x; i < S; i += 256) {
        float v = (p[i] - mean) * inv;
        p[i] = relu ? fmaxf(v, 0.f) : v;
    }
}

// ---------------------------------------------------------------------------
// BatchNorm-eval affine: dst = g[c]*BN_SCALE*src + be[c] (+relu)
// ---------------------------------------------------------------------------
__global__ void k_bn(float* dst, long dImg, const float* src, long sImg,
                     const float* g, const float* be, int C, int S, long total, int relu) {
    long idx = (long)blockIdx.x * blockDim.x + threadIdx.x;
    if (idx >= total) return;
    long CS = (long)C * S;
    long b = idx / CS;
    long r = idx - b * CS;
    int c = (int)(r / S);
    float v = src[b * sImg + r];
    v = g[c] * BN_SCALE_F * v + be[c];
    if (relu) v = fmaxf(v, 0.f);
    dst[b * dImg + r] = v;
}

// mode: 0 copy, 1 relu, 2 tanh, 3 sigmoid
__global__ void k_act(float* dst, long dImg, const float* src, long sImg,
                      long CS, long total, int mode) {
    long idx = (long)blockIdx.x * blockDim.x + threadIdx.x;
    if (idx >= total) return;
    long b = idx / CS;
    long r = idx - b * CS;
    float v = src[b * sImg + r];
    if (mode == 1) v = fmaxf(v, 0.f);
    else if (mode == 2) v = tanhf(v);
    else if (mode == 3) v = 1.f / (1.f + expf(-v));
    dst[b * dImg + r] = v;
}

__global__ void k_add_relu(float* dst, long dImg, const float* a, long aImg,
                           const float* b, long bImg, long CS, long total, int relu) {
    long idx = (long)blockIdx.x * blockDim.x + threadIdx.x;
    if (idx >= total) return;
    long bb = idx / CS;
    long r  = idx - bb * CS;
    float v = a[bb * aImg + r] + b[bb * bImg + r];
    if (relu) v = fmaxf(v, 0.f);
    dst[bb * dImg + r] = v;
}

__global__ void k_mul(float* dst, long dImg, const float* a, long aImg,
                      const float* b, long bImg, long CS, long total) {
    long idx = (long)blockIdx.x * blockDim.x + threadIdx.x;
    if (idx >= total) return;
    long bb = idx / CS;
    long r  = idx - bb * CS;
    dst[bb * dImg + r] = a[bb * aImg + r] * b[bb * bImg + r];
}

// net = (1-z)*net + z*q
__global__ void k_gru(float* net, long nImg, const float* z, long zImg,
                      const float* q, long qImg, long CS, long total) {
    long idx = (long)blockIdx.x * blockDim.x + threadIdx.x;
    if (idx >= total) return;
    long b = idx / CS;
    long r = idx - b * CS;
    float zz = z[b * zImg + r];
    float nn = net[b * nImg + r];
    net[b * nImg + r] = (1.f - zz) * nn + zz * q[b * qImg + r];
}

__global__ void k_zero(float* p, long n) {
    long idx = (long)blockIdx.x * blockDim.x + threadIdx.x;
    if (idx < n) p[idx] = 0.f;
}

__global__ void k_avgpool(const float* __restrict__ src, float* __restrict__ dst,
                          int NB, int h, int w) {
    int ho2 = h >> 1, wo2 = w >> 1;
    long idx = (long)blockIdx.x * blockDim.x + threadIdx.x;
    long total = (long)NB * ho2 * wo2;
    if (idx >= total) return;
    int wo = (int)(idx % wo2);
    int ho = (int)((idx / wo2) % ho2);
    long nb = idx / ((long)wo2 * ho2);
    const float* p = src + (nb * h + 2 * ho) * (long)w + 2 * wo;
    dst[idx] = 0.25f * (p[0] + p[1] + p[w] + p[w + 1]);
}

// ---------------------------------------------------------------------------
// Correlation lookup: out[B][324][S]; pyramid levels [B*S][Hl][Wl]
// ---------------------------------------------------------------------------
__global__ void k_corr_lookup(const float* __restrict__ p0, const float* __restrict__ p1,
                              const float* __restrict__ p2, const float* __restrict__ p3,
                              const float* __restrict__ flow, float* __restrict__ out,
                              int B, int H, int W) {
    const int S = H * W;
    long idx = (long)blockIdx.x * blockDim.x + threadIdx.x;
    long total = (long)B * 324 * S;
    if (idx >= total) return;
    int s  = (int)(idx % S);
    int ch = (int)((idx / S) % 324);
    int b  = (int)(idx / ((long)324 * S));
    int lvl = ch / 81;
    int t = ch - lvl * 81;
    int di = t / 9, dj = t - di * 9;
    float cx = (float)(s % W) + flow[((long)b * 2 + 0) * S + s];
    float cy = (float)(s / W) + flow[((long)b * 2 + 1) * S + s];
    float scale = 1.0f / (float)(1 << lvl);
    float x = cx * scale + (float)(di - 4);
    float y = cy * scale + (float)(dj - 4);
    int Hl = H >> lvl, Wl = W >> lvl;
    const float* base = (lvl == 0) ? p0 : (lvl == 1) ? p1 : (lvl == 2) ? p2 : p3;
    const float* img = base + ((long)b * S + s) * ((long)Hl * Wl);
    float x0f = floorf(x), y0f = floorf(y);
    float dx = x - x0f, dy = y - y0f;
    int x0 = (int)x0f, y0 = (int)y0f;
    float v = 0.f;
    if (x0 >= 0 && x0 < Wl && y0 >= 0 && y0 < Hl)         v += (1.f - dx) * (1.f - dy) * img[(long)y0 * Wl + x0];
    if (x0 + 1 >= 0 && x0 + 1 < Wl && y0 >= 0 && y0 < Hl) v += dx * (1.f - dy) * img[(long)y0 * Wl + x0 + 1];
    if (x0 >= 0 && x0 < Wl && y0 + 1 >= 0 && y0 + 1 < Hl) v += (1.f - dx) * dy * img[(long)(y0 + 1) * Wl + x0];
    if (x0 + 1 >= 0 && x0 + 1 < Wl && y0 + 1 >= 0 && y0 + 1 < Hl) v += dx * dy * img[(long)(y0 + 1) * Wl + x0 + 1];
    out[idx] = v;
}

// ---------------------------------------------------------------------------
// Convex upsample: flow [B][2][H][W], mask [B][576][H][W] -> out [B][2][8H][8W]
// ---------------------------------------------------------------------------
__global__ void k_upsample(const float* __restrict__ flow, const float* __restrict__ mask,
                           float* __restrict__ out, int B, int H, int W) {
    int H8 = 8 * H, W8 = 8 * W;
    long idx = (long)blockIdx.x * blockDim.x + threadIdx.x;
    long total = (long)B * 2 * H8 * W8;
    if (idx >= total) return;
    int w8 = (int)(idx % W8);
    int h8 = (int)((idx / W8) % H8);
    int c  = (int)((idx / ((long)W8 * H8)) % 2);
    int n  = (int)(idx / ((long)2 * W8 * H8));
    int h = h8 >> 3, i = h8 & 7, w = w8 >> 3, j = w8 & 7;
    float m[9];
    float mx = -1e30f;
#pragma unroll
    for (int k = 0; k < 9; ++k) {
        m[k] = mask[(((long)n * 576 + (k * 64 + i * 8 + j)) * H + h) * W + w];
        mx = fmaxf(mx, m[k]);
    }
    float sum = 0.f;
#pragma unroll
    for (int k = 0; k < 9; ++k) { m[k] = expf(m[k] - mx); sum += m[k]; }
    float v = 0.f;
#pragma unroll
    for (int k = 0; k < 9; ++k) {
        int kh = k / 3, kw = k - kh * 3;
        int hh = h + kh - 1, ww = w + kw - 1;
        float f = (hh >= 0 && hh < H && ww >= 0 && ww < W)
                      ? 8.f * flow[(((long)n * 2 + c) * H + hh) * W + ww] : 0.f;
        v += m[k] * f;
    }
    out[idx] = v / sum;
}

// ===========================================================================
// Host side
// ===========================================================================
struct ConvP { const float* w; const float* b; };
struct NormP { const float* g; const float* be; };
struct BlockP { ConvP c1, c2, down; NormP n1, n2, n3; bool has_down; };
struct EncP  { ConvP conv1, conv2; NormP n1; BlockP blk[6]; };
struct UpdP  { ConvP convc1, convc2, convf1, convf2, convm,
                     convq1, convq2, convr1, convr2, convz1, convz2,
                     fh1, fh2, mk1, mk2; };
struct AllP  { const float* img1; const float* img2; EncP fnet, cnet; UpdP upd; };

struct Cursor {
    void* const* din; const int* sizes; int n; int i; bool ok;
    const float* take(long expect) {
        if (i >= n) { ok = false; return nullptr; }
        if ((long)sizes[i] != expect) ok = false;
        return (const float*)din[i++];
    }
};

static void readConv(Cursor& c, ConvP& p, int o, int ci, int kh, int kw, bool wFirst) {
    long ws = (long)o * ci * kh * kw;
    if (wFirst) { p.w = c.take(ws); p.b = c.take(o); }
    else        { p.b = c.take(o);  p.w = c.take(ws); }
}
static void readNorm(Cursor& c, NormP& p, int nCh, bool gFirst) {
    if (gFirst) { p.g = c.take(nCh); p.be = c.take(nCh); }
    else        { p.be = c.take(nCh); p.g = c.take(nCh); }
}
static void readBlock(Cursor& c, BlockP& b, int ci, int co, int stride, bool bn, bool ins) {
    readConv(c, b.c1, co, ci, 3, 3, ins);
    readConv(c, b.c2, co, co, 3, 3, ins);
    b.has_down = (stride != 1);
    if (ins) { // insertion order: c1,c2,[n1,n2],[down],[n3]
        if (bn) { readNorm(c, b.n1, co, true); readNorm(c, b.n2, co, true); }
        if (stride != 1) {
            readConv(c, b.down, co, ci, 1, 1, true);
            if (bn) readNorm(c, b.n3, co, true);
        }
    } else { // sorted: c1,c2,down,n1,n2,n3
        if (stride != 1) readConv(c, b.down, co, ci, 1, 1, false);
        if (bn) {
            readNorm(c, b.n1, co, false); readNorm(c, b.n2, co, false);
            if (stride != 1) readNorm(c, b.n3, co, false);
        }
    }
}
static const int BLK_CI[6] = {64, 64, 64, 96, 96, 128};
static const int BLK_CO[6] = {64, 64, 96, 96, 128, 128};
static const int BLK_ST[6] = {1, 1, 2, 1, 2, 1};
static void readEnc(Cursor& c, EncP& e, bool bn, bool ins) {
    if (ins) { // conv1, blocks, conv2, [n1]
        readConv(c, e.conv1, 64, 3, 7, 7, true);
        for (int i = 0; i < 6; ++i) readBlock(c, e.blk[i], BLK_CI[i], BLK_CO[i], BLK_ST[i], bn, true);
        readConv(c, e.conv2, 256, 128, 1, 1, true);
        if (bn) readNorm(c, e.n1, 64, true);
    } else { // sorted: conv1, conv2, blocks, [n1]
        readConv(c, e.conv1, 64, 3, 7, 7, false);
        readConv(c, e.conv2, 256, 128, 1, 1, false);
        for (int i = 0; i < 6; ++i) readBlock(c, e.blk[i], BLK_CI[i], BLK_CO[i], BLK_ST[i], bn, false);
        if (bn) readNorm(c, e.n1, 64, false);
    }
}
static void readUpd(Cursor& c, UpdP& u, bool ins) {
    if (ins) { // insertion: c1,c2,f1,f2,m,z1,r1,q1,z2,r2,q2,fh1,fh2,mk1,mk2
        readConv(c, u.convc1, 256, 324, 1, 1, true);
        readConv(c, u.convc2, 192, 256, 3, 3, true);
        readConv(c, u.convf1, 128, 2, 7, 7, true);
        readConv(c, u.convf2, 64, 128, 3, 3, true);
        readConv(c, u.convm, 126, 256, 3, 3, true);
        readConv(c, u.convz1, 128, 384, 1, 5, true);
        readConv(c, u.convr1, 128, 384, 1, 5, true);
        readConv(c, u.convq1, 128, 384, 1, 5, true);
        readConv(c, u.convz2, 128, 384, 5, 1, true);
        readConv(c, u.convr2, 128, 384, 5, 1, true);
        readConv(c, u.convq2, 128, 384, 5, 1, true);
        readConv(c, u.fh1, 256, 128, 3, 3, true);
        readConv(c, u.fh2, 2, 256, 3, 3, true);
        readConv(c, u.mk1, 256, 128, 3, 3, true);
        readConv(c, u.mk2, 576, 256, 1, 1, true);
    } else { // sorted keys
        readConv(c, u.convc1, 256, 324, 1, 1, false);
        readConv(c, u.convc2, 192, 256, 3, 3, false);
        readConv(c, u.convf1, 128, 2, 7, 7, false);
        readConv(c, u.convf2, 64, 128, 3, 3, false);
        readConv(c, u.convm, 126, 256, 3, 3, false);
        readConv(c, u.convq1, 128, 384, 1, 5, false);
        readConv(c, u.convq2, 128, 384, 5, 1, false);
        readConv(c, u.convr1, 128, 384, 1, 5, false);
        readConv(c, u.convr2, 128, 384, 5, 1, false);
        readConv(c, u.convz1, 128, 384, 1, 5, false);
        readConv(c, u.convz2, 128, 384, 5, 1, false);
        readConv(c, u.fh1, 256, 128, 3, 3, false);
        readConv(c, u.fh2, 2, 256, 3, 3, false);
        readConv(c, u.mk1, 256, 128, 3, 3, false);
        readConv(c, u.mk2, 576, 256, 1, 1, false);
    }
}
static bool readAll(void* const* din, const int* sizes, int n, bool ins, AllP& P) {
    Cursor c{din, sizes, n, 0, true};
    P.img1 = c.take((long)2 * 3 * 384 * 512);
    P.img2 = c.take((long)2 * 3 * 384 * 512);
    if (ins) {
        readEnc(c, P.fnet, false, true);
        readEnc(c, P.cnet, true, true);
        readUpd(c, P.upd, true);
    } else {
        readEnc(c, P.cnet, true, false);
        readEnc(c, P.fnet, false, false);
        readUpd(c, P.upd, false);
    }
    return c.ok && c.i == n;
}

static inline int ceilDiv(long a, long b) { return (int)((a + b - 1) / b); }
static inline int pad16(int x) { return (x + 15) & ~15; }
static inline int pad32(int x) { return (x + 31) & ~31; }

static void pack_weights(hipStream_t st, const float* w, _Float16* A, int M, int K) {
    int Mp = pad16(M), Kp = pad32(K);
    long tot = (long)Mp * Kp;
    k_pack_w<<<ceilDiv(tot, 256), 256, 0, st>>>(w, A, M, K, Mp, Kp);
}

// Implicit-GEMM conv for B images (host loop over images, shared im2col buffer)
static void conv_gemm(hipStream_t st, const float* in, long inImgStride, int B,
                      int Cin, int Hin, int Win, const _Float16* Apack, const float* bias,
                      int Cout, int KH, int KW, int stride, int padH, int padW,
                      float* outBase, long outImgStride, _Float16* btbuf,
                      float alpha, int relu) {
    int Hout = (Hin + 2 * padH - KH) / stride + 1;
    int Wout = (Win + 2 * padW - KW) / stride + 1;
    int K = Cin * KH * KW, Kp = pad32(K), Mp = pad16(Cout);
    int Nsp = Hout * Wout;
    long tot = (long)Nsp * Kp;
    dim3 gg(ceilDiv(Nsp, 128), ceilDiv(Mp, 64), 1);
    for (int img = 0; img < B; ++img) {
        k_im2col<<<ceilDiv(tot, 256), 256, 0, st>>>(in + (long)img * inImgStride,
                                                    Cin, Hin, Win, KH, KW, stride, padH, padW,
                                                    Hout, Wout, btbuf, Kp);
        k_gemm<<<gg, 256, 0, st>>>(Apack, btbuf, outBase + (long)img * outImgStride,
                                   bias, Mp, Kp, Nsp, Cout, alpha, relu);
    }
}

static void norm_apply(hipStream_t st, float* x, int C, int S, int B, bool bn,
                       const NormP& np, int relu) {
    if (bn) {
        long total = (long)B * C * S;
        k_bn<<<ceilDiv(total, 256), 256, 0, st>>>(x, (long)C * S, x, (long)C * S,
                                                  np.g, np.be, C, S, total, relu);
    } else {
        k_instnorm<<<dim3(C, B), 256, 0, st>>>(x, (long)C * S, S, relu);
    }
}

static void run_encoder(hipStream_t st, const float* input, int B, const EncP& P, bool bn,
                        float* out, float* Y, float* H1, float* H2,
                        _Float16* wpack, _Float16* btbuf) {
    // conv1: 7x7 s2 p3, 3 -> 64
    pack_weights(st, P.conv1.w, wpack, 64, 3 * 49);
    conv_gemm(st, input, (long)3 * 384 * 512, B, 3, 384, 512, wpack, P.conv1.b,
              64, 7, 7, 2, 3, 3, Y, (long)64 * 192 * 256, btbuf, 1.f, 0);
    int C = 64, Hc = 192, Wc = 256;
    norm_apply(st, Y, C, Hc * Wc, B, bn, P.n1, 1);

    for (int bi = 0; bi < 6; ++bi) {
        const BlockP& blk = P.blk[bi];
        int Co = BLK_CO[bi], st2 = BLK_ST[bi];
        int Ho = (Hc + 2 - 3) / st2 + 1;
        int Wo = (Wc + 2 - 3) / st2 + 1;
        long S  = (long)Hc * Wc;
        long So = (long)Ho * Wo;
        // c1
        pack_weights(st, blk.c1.w, wpack, Co, C * 9);
        conv_gemm(st, Y, (long)C * S, B, C, Hc, Wc, wpack, blk.c1.b,
                  Co, 3, 3, st2, 1, 1, H1, (long)Co * So, btbuf, 1.f, 0);
        norm_apply(st, H1, Co, (int)So, B, bn, blk.n1, 1);
        // c2
        pack_weights(st, blk.c2.w, wpack, Co, Co * 9);
        conv_gemm(st, H1, (long)Co * So, B, Co, Ho, Wo, wpack, blk.c2.b,
                  Co, 3, 3, 1, 1, 1, H2, (long)Co * So, btbuf, 1.f, 0);
        norm_apply(st, H2, Co, (int)So, B, bn, blk.n2, 1);
        long total = (long)B * Co * So;
        if (st2 != 1) {
            pack_weights(st, blk.down.w, wpack, Co, C);
            conv_gemm(st, Y, (long)C * S, B, C, Hc, Wc, wpack, blk.down.b,
                      Co, 1, 1, st2, 0, 0, H1, (long)Co * So, btbuf, 1.f, 0);
            norm_apply(st, H1, Co, (int)So, B, bn, blk.n3, 0);
            k_add_relu<<<ceilDiv(total, 256), 256, 0, st>>>(Y, (long)Co * So,
                H1, (long)Co * So, H2, (long)Co * So, (long)Co * So, total, 1);
        } else {
            k_add_relu<<<ceilDiv(total, 256), 256, 0, st>>>(Y, (long)Co * So,
                Y, (long)Co * So, H2, (long)Co * So, (long)Co * So, total, 1);
        }
        C = Co; Hc = Ho; Wc = Wo;
    }
    // conv2: 1x1, 128 -> 256, bias only
    long S = (long)Hc * Wc;
    pack_weights(st, P.conv2.w, wpack, 256, 128);
    conv_gemm(st, Y, (long)C * S, B, C, Hc, Wc, wpack, P.conv2.b,
              256, 1, 1, 1, 0, 0, out, (long)256 * S, btbuf, 1.f, 0);
}

extern "C" void kernel_launch(void* const* d_in, const int* in_sizes, int n_in,
                              void* d_out, int out_size, void* d_ws, size_t ws_size,
                              hipStream_t stream) {
    (void)out_size; (void)ws_size;
    AllP P;
    if (!readAll(d_in, in_sizes, n_in, false, P))
        if (!readAll(d_in, in_sizes, n_in, true, P))
            readAll(d_in, in_sizes, n_in, false, P); // fallback: sorted

    const int B = 2, H = 48, W = 64, S = H * W;
    const long imgPix = (long)3 * 384 * 512;

    // ---- workspace layout -------------------------------------------------
    size_t off = 0;
    auto alloc = [&](size_t bytes) -> void* {
        void* p = (char*)d_ws + off;
        off += (bytes + 255) & ~(size_t)255;
        return p;
    };
    float* x0   = (float*)alloc(4 * imgPix * 4);
    float* Y    = (float*)alloc((size_t)4 * 64 * 192 * 256 * 4);
    float* Hb1  = (float*)alloc((size_t)4 * 64 * 192 * 256 * 4);
    float* Hb2  = (float*)alloc((size_t)4 * 64 * 192 * 256 * 4);
    float* ff   = (float*)alloc((size_t)4 * 256 * S * 4);   // fnet out [4,256,S]
    float* cc   = (float*)alloc((size_t)2 * 256 * S * 4);   // cnet out
    _Float16* At  = (_Float16*)alloc((size_t)2 * S * 256 * 2);
    _Float16* Btc = (_Float16*)alloc((size_t)2 * S * 256 * 2);
    float* pyr0 = (float*)alloc((size_t)2 * S * S * 4);
    float* pyr1 = (float*)alloc((size_t)2 * S * (S / 4) * 4);
    float* pyr2 = (float*)alloc((size_t)2 * S * (S / 16) * 4);
    float* pyr3 = (float*)alloc((size_t)2 * S * (S / 64) * 4);
    _Float16* btbuf = (_Float16*)alloc((size_t)49152 * 576 * 2); // max im2col
    _Float16* wpack = (_Float16*)alloc((size_t)256 * 2304 * 2);  // encoder packs
    float* hx    = (float*)alloc((size_t)B * 384 * S * 4);
    float* qin   = (float*)alloc((size_t)B * 384 * S * 4);
    float* corrf = (float*)alloc((size_t)B * 324 * S * 4);
    float* corb  = (float*)alloc((size_t)B * 256 * S * 4);
    float* motin = (float*)alloc((size_t)B * 256 * S * 4);
    float* flo1  = (float*)alloc((size_t)B * 128 * S * 4);
    float* zbuf  = (float*)alloc((size_t)B * 128 * S * 4);
    float* rbuf  = (float*)alloc((size_t)B * 128 * S * 4);
    float* qbuf  = (float*)alloc((size_t)B * 128 * S * 4);
    float* t256  = (float*)alloc((size_t)B * 256 * S * 4);
    float* dflow = (float*)alloc((size_t)B * 2 * S * 4);
    float* maskb = (float*)alloc((size_t)B * 576 * S * 4);
    float* facc  = (float*)alloc((size_t)B * 2 * S * 4);

    // persistent packed update-conv weights
    struct UD { const ConvP* c; int Cout, Cin, KH, KW; _Float16* pack; };
    UD ud[15] = {
        {&P.upd.convc1, 256, 324, 1, 1, nullptr}, {&P.upd.convc2, 192, 256, 3, 3, nullptr},
        {&P.upd.convf1, 128, 2, 7, 7, nullptr},   {&P.upd.convf2, 64, 128, 3, 3, nullptr},
        {&P.upd.convm, 126, 256, 3, 3, nullptr},
        {&P.upd.convz1, 128, 384, 1, 5, nullptr}, {&P.upd.convr1, 128, 384, 1, 5, nullptr},
        {&P.upd.convq1, 128, 384, 1, 5, nullptr},
        {&P.upd.convz2, 128, 384, 5, 1, nullptr}, {&P.upd.convr2, 128, 384, 5, 1, nullptr},
        {&P.upd.convq2, 128, 384, 5, 1, nullptr},
        {&P.upd.fh1, 256, 128, 3, 3, nullptr},    {&P.upd.fh2, 2, 256, 3, 3, nullptr},
        {&P.upd.mk1, 256, 128, 3, 3, nullptr},    {&P.upd.mk2, 576, 256, 1, 1, nullptr}};
    for (int i = 0; i < 15; ++i) {
        int K = ud[i].Cin * ud[i].KH * ud[i].KW;
        ud[i].pack = (_Float16*)alloc((size_t)pad16(ud[i].Cout) * pad32(K) * 2);
        pack_weights(stream, ud[i].c->w, ud[i].pack, ud[i].Cout, K);
    }
    auto uconv = [&](const UD& d, const float* in, long inImg, int stride2, int pH, int pW,
                     float* outB, long outImg, float alpha, int relu) {
        conv_gemm(stream, in, inImg, B, d.Cin, H, W, d.pack, d.c->b,
                  d.Cout, d.KH, d.KW, stride2, pH, pW, outB, outImg, btbuf, alpha, relu);
    };

    // ---- encoders ---------------------------------------------------------
    hipMemcpyAsync(x0, P.img1, 2 * imgPix * 4, hipMemcpyDeviceToDevice, stream);
    hipMemcpyAsync(x0 + 2 * imgPix, P.img2, 2 * imgPix * 4, hipMemcpyDeviceToDevice, stream);
    run_encoder(stream, x0, 4, P.fnet, false, ff, Y, Hb1, Hb2, wpack, btbuf);
    run_encoder(stream, P.img1, 2, P.cnet, true, cc, Y, Hb1, Hb2, wpack, btbuf);

    // net = tanh(cc[:, :128]) -> hx ch0..128 ; inp = relu(cc[:,128:]) -> hx ch128..256
    {
        long tot = (long)B * 128 * S;
        k_act<<<ceilDiv(tot, 256), 256, 0, stream>>>(hx, (long)384 * S, cc, (long)256 * S,
                                                     (long)128 * S, tot, 2);
        k_act<<<ceilDiv(tot, 256), 256, 0, stream>>>(hx + (long)128 * S, (long)384 * S,
                                                     cc + (long)128 * S, (long)256 * S,
                                                     (long)128 * S, tot, 3 - 2); // mode 1 relu
    }

    // ---- correlation pyramid ---------------------------------------------
    for (int b = 0; b < 2; ++b) {
        long tot = (long)S * 256;
        k_pack_feat<<<ceilDiv(tot, 256), 256, 0, stream>>>(ff + (long)b * 256 * S,
                                                           At + (long)b * S * 256, S, 256);
        k_pack_feat<<<ceilDiv(tot, 256), 256, 0, stream>>>(ff + (long)(2 + b) * 256 * S,
                                                           Btc + (long)b * S * 256, S, 256);
    }
    for (int b = 0; b < 2; ++b) {
        dim3 gg(ceilDiv(S, 128), ceilDiv(S, 64), 1);
        k_gemm<<<gg, 256, 0, stream>>>(At + (long)b * S * 256, Btc + (long)b * S * 256,
                                       pyr0 + (long)b * S * S, nullptr,
                                       S, 256, S, S, 0.0625f, 0);
    }
    {
        int NB = 2 * S;
        k_avgpool<<<ceilDiv((long)NB * 24 * 32, 256), 256, 0, stream>>>(pyr0, pyr1, NB, 48, 64);
        k_avgpool<<<ceilDiv((long)NB * 12 * 16, 256), 256, 0, stream>>>(pyr1, pyr2, NB, 24, 32);
        k_avgpool<<<ceilDiv((long)NB * 6 * 8, 256), 256, 0, stream>>>(pyr2, pyr3, NB, 12, 16);
    }

    // ---- iterative updates ------------------------------------------------
    k_zero<<<ceilDiv((long)B * 2 * S, 256), 256, 0, stream>>>(facc, (long)B * 2 * S);
    const long I384 = (long)384 * S, I256 = (long)256 * S, I128 = (long)128 * S, I2 = (long)2 * S;
    const long CS128 = I128, CS256 = I256, CS2 = I2;
    const long T128 = (long)B * 128 * S, T256 = (long)B * 256 * S, T2 = (long)B * 2 * S;

    for (int it = 0; it < 12; ++it) {
        // correlation lookup
        {
            long tot = (long)B * 324 * S;
            k_corr_lookup<<<ceilDiv(tot, 256), 256, 0, stream>>>(pyr0, pyr1, pyr2, pyr3,
                                                                 facc, corrf, B, H, W);
        }
        // motion encoder
        uconv(ud[0], corrf, (long)324 * S, 1, 0, 0, corb, I256, 1.f, 1);             // convc1 1x1
        uconv(ud[1], corb, I256, 1, 1, 1, motin, I256, 1.f, 1);                      // convc2 -> motin ch0..192
        uconv(ud[2], facc, I2, 1, 3, 3, flo1, I128, 1.f, 1);                         // convf1 7x7
        uconv(ud[3], flo1, I128, 1, 1, 1, motin + (long)192 * S, I256, 1.f, 1);      // convf2 -> ch192..256
        uconv(ud[4], motin, I256, 1, 1, 1, hx + (long)256 * S, I384, 1.f, 1);        // convm -> hx ch256..382
        k_act<<<ceilDiv(T2, 256), 256, 0, stream>>>(hx + (long)382 * S, I384, facc, I2, CS2, T2, 0);

        // GRU, two separable stages
        for (int stage = 0; stage < 2; ++stage) {
            const UD& Uz = ud[5 + stage * 3 + 0];
            const UD& Ur = ud[5 + stage * 3 + 1];
            const UD& Uq = ud[5 + stage * 3 + 2];
            int pH = (stage == 0) ? 0 : 2, pW = (stage == 0) ? 2 : 0;
            uconv(Uz, hx, I384, 1, pH, pW, zbuf, I128, 1.f, 0);
            k_act<<<ceilDiv(T128, 256), 256, 0, stream>>>(zbuf, I128, zbuf, I128, CS128, T128, 3);
            uconv(Ur, hx, I384, 1, pH, pW, rbuf, I128, 1.f, 0);
            k_act<<<ceilDiv(T128, 256), 256, 0, stream>>>(rbuf, I128, rbuf, I128, CS128, T128, 3);
            k_mul<<<ceilDiv(T128, 256), 256, 0, stream>>>(qin, I384, rbuf, I128, hx, I384, CS128, T128);
            k_act<<<ceilDiv(T256, 256), 256, 0, stream>>>(qin + I128, I384, hx + I128, I384, CS256, T256, 0);
            uconv(Uq, qin, I384, 1, pH, pW, qbuf, I128, 1.f, 0);
            k_act<<<ceilDiv(T128, 256), 256, 0, stream>>>(qbuf, I128, qbuf, I128, CS128, T128, 2);
            k_gru<<<ceilDiv(T128, 256), 256, 0, stream>>>(hx, I384, zbuf, I128, qbuf, I128, CS128, T128);
        }

        // flow head
        uconv(ud[11], hx, I384, 1, 1, 1, t256, I256, 1.f, 1);   // fh1
        uconv(ud[12], t256, I256, 1, 1, 1, dflow, I2, 1.f, 0);  // fh2
        if (it == 11) {
            uconv(ud[13], hx, I384, 1, 1, 1, t256, I256, 1.f, 1);    // mk1
            uconv(ud[14], t256, I256, 1, 0, 0, maskb, (long)576 * S, 0.25f, 0); // mk2, 0.25*
        }
        k_add_relu<<<ceilDiv(T2, 256), 256, 0, stream>>>(facc, I2, facc, I2, dflow, I2, CS2, T2, 0);
    }

    // ---- upsample ---------------------------------------------------------
    {
        long tot = (long)B * 2 * (8 * H) * (8 * W);
        k_upsample<<<ceilDiv(tot, 256), 256, 0, stream>>>(facc, maskb, (float*)d_out, B, H, W);
    }
}
